// RNNFeaturizer_63136019251203
// MI455X (gfx1250) — matrix operand order repaired and checked by reading, not compiled, not run on hardware
//
#include <hip/hip_runtime.h>
#include <hip/hip_bf16.h>

#define NINP 64
#define NHID 4096
#define TT   256
#define BB   32

typedef __attribute__((ext_vector_type(16))) __bf16       v16bf;
typedef __attribute__((ext_vector_type(8)))  float        v8f;
typedef __attribute__((ext_vector_type(4)))  unsigned int u32x4;

union FragBF { v16bf v; u32x4 u[2]; };

__device__ __forceinline__ unsigned short f2bf(float x) {
    union { float f; unsigned u; } v; v.f = x;
    unsigned r = v.u + 0x7FFFu + ((v.u >> 16) & 1u);   // round-to-nearest-even
    return (unsigned short)(r >> 16);
}
__device__ __forceinline__ unsigned pack2(float a, float b) {
    return (unsigned)f2bf(a) | ((unsigned)f2bf(b) << 16);
}
__device__ __forceinline__ float fsig(float x)  { return 1.0f / (1.0f + __expf(-x)); }
__device__ __forceinline__ float ftanh(float x) { return 1.0f - 2.0f / (__expf(2.0f * x) + 1.0f); }

// ---- Pack W_hh (fp32 [16384,4096]) into gate-interleaved B-fragment order, bf16 ----
// layout: [group(256)][kc(128)][gate(4)][lane(32)][8 dwords]
// B layout (16-bit 32x16): lane n=l&15, k = kc*32 + (q>>2)*16 + (l>>4)*8 + (q&3)*2 + {0,1}
// row j = gate*4096 + group*16 + n
__global__ void pack_whh(const float* __restrict__ W, unsigned* __restrict__ out) {
    unsigned d = blockIdx.x * 256u + threadIdx.x;          // < 33554432
    unsigned group = d >> 17;                              // 131072 dwords per group
    unsigned r  = d & 131071u;
    unsigned kc = r >> 10;                                 // 1024 dwords per kc
    unsigned r2 = r & 1023u;
    unsigned g  = r2 >> 8, l = (r2 >> 3) & 31u, q = r2 & 7u;
    unsigned j  = g * NHID + group * 16u + (l & 15u);
    unsigned k  = kc * 32u + ((q >> 2) * 16u) + ((l >> 4) * 8u) + (q & 3u) * 2u;
    const float* p = W + (size_t)j * NHID + k;
    out[d] = pack2(p[0], p[1]);
}

// ---- Pack W_ih (fp32 [16384,64]) the same way: [group][kc(2)][gate][lane][8dw] ----
__global__ void pack_wih(const float* __restrict__ W, unsigned* __restrict__ out) {
    unsigned d = blockIdx.x * 256u + threadIdx.x;          // < 524288
    unsigned group = d >> 11;                              // 2048 dwords per group
    unsigned r  = d & 2047u;
    unsigned kc = r >> 10;
    unsigned r2 = r & 1023u;
    unsigned g  = r2 >> 8, l = (r2 >> 3) & 31u, q = r2 & 7u;
    unsigned j  = g * NHID + group * 16u + (l & 15u);
    unsigned k  = kc * 32u + ((q >> 2) * 16u) + ((l >> 4) * 8u) + (q & 3u) * 2u;
    const float* p = W + (size_t)j * NINP + k;
    out[d] = pack2(p[0], p[1]);
}

// ---- Pack embeddings for every timestep into A-fragment order ----
// A layout (16-bit 16x32): lane m=l&15, k = (q>>2)*16 + (l>>4)*8 + (q&3)*2 + {0,1}
// layout: [t][mt][kc][lane][8 dwords]
__global__ void pack_x(const int* __restrict__ tokens, const float* __restrict__ emb,
                       unsigned* __restrict__ out) {
    unsigned d = blockIdx.x * 256u + threadIdx.x;          // < 262144
    unsigned t = d >> 10, r = d & 1023u;
    unsigned mt = r >> 9, kc = (r >> 8) & 1u, l = (r >> 3) & 31u, q = r & 7u;
    unsigned m = (l & 15u) + 16u * mt;
    unsigned k = kc * 32u + ((q >> 2) * 16u) + ((l >> 4) * 8u) + (q & 3u) * 2u;
    int tok = tokens[t * BB + m];
    const float* p = emb + (size_t)tok * NINP + k;
    out[d] = pack2(p[0], p[1]);
}

__global__ void bias_fuse(const float* __restrict__ a, const float* __restrict__ b,
                          float* __restrict__ o) {
    int i = blockIdx.x * 256 + threadIdx.x;
    o[i] = a[i] + b[i];
}

__global__ void zero_f(float* __restrict__ p) {
    p[blockIdx.x * 256 + threadIdx.x] = 0.0f;
}

// ---- One LSTM timestep. 256 waves; wave `group` owns hidden cols [group*16, group*16+16) ----
__global__ void __launch_bounds__(128)
lstm_step(const u32x4* __restrict__ Wp,        // W_hh packed (uint4 units)
          const u32x4* __restrict__ WIp,       // W_ih packed
          const u32x4* __restrict__ Xp,        // X packed, all timesteps
          const float* __restrict__ bias,
          const u32x4* __restrict__ h_in,      // bf16 row-major [32][4096] viewed as uint4
          unsigned short* __restrict__ h_out,  // bf16 row-major [32][4096]
          float* __restrict__ c,               // fp32 [32][4096]
          float* __restrict__ feat,            // d_out fp32 [32][4096]
          const int* __restrict__ seq_len,
          int t) {
    const unsigned lane  = threadIdx.x & 31u;
    const unsigned group = blockIdx.x * (blockDim.x >> 5) + (threadIdx.x >> 5); // 0..255
    const unsigned n     = lane & 15u;

    v8f acc[4][2] = {};   // [gate][batch-half], 16x16 f32 tiles

    // ===== h_{t-1} @ W_hh^T, K = 4096 in chunks of 32 =====
    const unsigned kb_lo = (lane >> 4) * 8u;
    // all 4 gate fragments for (group,kc) are 4 consecutive KiB from one base
    const u32x4* wbase = Wp + (size_t)group * (128u * 256u) + lane * 2u;
    #pragma unroll 1
    for (unsigned kc = 0; kc < 128u; ++kc) {
        FragBF A[2], Bg;
        unsigned kb = kc * 32u + kb_lo;
        #pragma unroll
        for (int mt = 0; mt < 2; ++mt) {
            const u32x4* hp = h_in + (size_t)(n + 16u * mt) * (NHID / 8) + (kb >> 3);
            A[mt].u[0] = hp[0];
            A[mt].u[1] = hp[2];                     // +16 bf16 = +2 uint4
        }
        const u32x4* bp = wbase + (size_t)kc * 256u;
        #pragma unroll
        for (int g = 0; g < 4; ++g) {
            Bg.u[0] = bp[g * 64];                   // gate stride = 1 KiB (imm offset)
            Bg.u[1] = bp[g * 64 + 1];
            acc[g][0] = __builtin_amdgcn_wmma_f32_16x16x32_bf16(
                false, A[0].v, false, Bg.v, (short)0, acc[g][0], false, false);
            acc[g][1] = __builtin_amdgcn_wmma_f32_16x16x32_bf16(
                false, A[1].v, false, Bg.v, (short)0, acc[g][1], false, false);
        }
    }

    // ===== x_t @ W_ih^T, K = 64 (2 chunks) =====
    const u32x4* wibase = WIp + (size_t)group * (2u * 256u) + lane * 2u;
    #pragma unroll
    for (unsigned kc = 0; kc < 2u; ++kc) {
        FragBF A[2], Bg;
        #pragma unroll
        for (int mt = 0; mt < 2; ++mt) {
            const u32x4* xp = Xp + ((((size_t)t * 2u + mt) * 2u + kc) * 32u + lane) * 2u;
            A[mt].u[0] = xp[0]; A[mt].u[1] = xp[1];
        }
        const u32x4* bp = wibase + (size_t)kc * 256u;
        #pragma unroll
        for (int g = 0; g < 4; ++g) {
            Bg.u[0] = bp[g * 64];
            Bg.u[1] = bp[g * 64 + 1];
            acc[g][0] = __builtin_amdgcn_wmma_f32_16x16x32_bf16(
                false, A[0].v, false, Bg.v, (short)0, acc[g][0], false, false);
            acc[g][1] = __builtin_amdgcn_wmma_f32_16x16x32_bf16(
                false, A[1].v, false, Bg.v, (short)0, acc[g][1], false, false);
        }
    }

    // ===== pointwise LSTM cell in registers; tiles share (lane, vgpr)->(m,n) map =====
    float bsv[4];
    #pragma unroll
    for (int g = 0; g < 4; ++g) bsv[g] = bias[(size_t)g * NHID + group * 16u + n];

    const unsigned col   = group * 16u + n;
    const unsigned mbase = (lane >> 4) << 3;          // C/D layout: m = r + 8*(lane>=16)
    #pragma unroll
    for (int mt = 0; mt < 2; ++mt) {
        #pragma unroll
        for (int r = 0; r < 8; ++r) {
            unsigned m  = mbase + (unsigned)r + 16u * mt;
            float iv = fsig (acc[0][mt][r] + bsv[0]);
            float fv = fsig (acc[1][mt][r] + bsv[1]);
            float gv = ftanh(acc[2][mt][r] + bsv[2]);
            float ov = fsig (acc[3][mt][r] + bsv[3]);
            size_t idx = (size_t)m * NHID + col;
            float cn = fv * c[idx] + iv * gv;
            c[idx] = cn;
            h_out[idx] = f2bf(ov * ftanh(cn));
            if (t < seq_len[m] || t == 0) feat[idx] = cn;   // matches reference masking
        }
    }
}

extern "C" void kernel_launch(void* const* d_in, const int* in_sizes, int n_in,
                              void* d_out, int out_size, void* d_ws, size_t ws_size,
                              hipStream_t stream) {
    (void)in_sizes; (void)n_in; (void)out_size; (void)ws_size;
    const int*   tokens  = (const int*)d_in[0];
    const int*   seq_len = (const int*)d_in[1];
    const float* emb     = (const float*)d_in[2];
    const float* W_ih    = (const float*)d_in[3];
    const float* W_hh    = (const float*)d_in[4];
    const float* b_ih    = (const float*)d_in[5];
    const float* b_hh    = (const float*)d_in[6];

    char* ws = (char*)d_ws;
    size_t off = 0;
    unsigned* Wp  = (unsigned*)(ws + off); off += 134217728;   // W_hh bf16 packed, 128 MiB (L2-resident)
    unsigned* WIp = (unsigned*)(ws + off); off += 2097152;     // W_ih bf16 packed, 2 MiB
    unsigned* Xp  = (unsigned*)(ws + off); off += 1048576;     // X fragments all t, 1 MiB
    float*   bias = (float*)   (ws + off); off += 65536;       // fused bias
    char*    hz   = ws + off;                                  // h0|h1|c, zero-init region
    unsigned short* h0 = (unsigned short*)hz;                  // 256 KiB
    unsigned short* h1 = (unsigned short*)(hz + 262144);       // 256 KiB
    float*          cc = (float*)(hz + 524288);                // 512 KiB
    float*        feat = (float*)d_out;

    pack_whh <<<131072, 256, 0, stream>>>(W_hh, Wp);
    pack_wih <<<2048,   256, 0, stream>>>(W_ih, WIp);
    pack_x   <<<1024,   256, 0, stream>>>(tokens, emb, Xp);
    bias_fuse<<<64,     256, 0, stream>>>(b_ih, b_hh, bias);
    zero_f   <<<1024,   256, 0, stream>>>((float*)hz);         // 262144 floats = h0+h1+c

    for (int t = 0; t < TT; ++t) {
        const unsigned short* hin = (t & 1) ? h1 : h0;
        unsigned short*      hout = (t & 1) ? h0 : h1;
        lstm_step<<<64, 128, 0, stream>>>((const u32x4*)Wp, (const u32x4*)WIp,
                                          (const u32x4*)Xp, bias,
                                          (const u32x4*)hin, hout, cc, feat, seq_len, t);
    }
}
